// MAB_1975684956570
// MI455X (gfx1250) — matrix operand order repaired and checked by reading, not compiled, not run on hardware
//
#include <hip/hip_runtime.h>
#include <hip/hip_bf16.h>

typedef _Float16 h16;
typedef __attribute__((ext_vector_type(16))) _Float16 v16h;
typedef __attribute__((ext_vector_type(8)))  _Float16 v8h;
typedef __attribute__((ext_vector_type(8)))  float    v8f;

// ---------------------------------------------------------------------------
// WMMA helpers (CDNA5 gfx1250, wave32). Layouts per CDNA5 ISA 7.12.2.
// ---------------------------------------------------------------------------
__device__ __forceinline__ v8f wmma_f16(v16h a, v16h b, v8f c) {
  // (neg_a, A, neg_b, B, c_mod, C, reuse_a, reuse_b)
  return __builtin_amdgcn_wmma_f32_16x16x32_f16(false, a, false, b, (short)0, c,
                                                false, false);
}

// A fragment: 16x32 (MxK) f16, element (m,k) at src[m*ld + k] (row-major).
// Per lane (m = lane&15, hi = lane>=16) the required K values are two
// contiguous 8-half runs: [8*hi .. 8*hi+7] and [16+8*hi .. 16+8*hi+7]
// -> two 16-byte vector loads. Caller guarantees 16B alignment.
__device__ __forceinline__ v16h load_a_frag(const h16* src, int ld) {
  const int lane = threadIdx.x & 31;
  const int m  = lane & 15;
  const int hi = (lane >> 4) & 1;
  const h16* p = src + (size_t)m * ld + hi * 8;
  union { v16h v; v8h h[2]; } u;
  u.h[0] = *reinterpret_cast<const v8h*>(p);        // K = 8*hi + 0..7
  u.h[1] = *reinterpret_cast<const v8h*>(p + 16);   // K = 16 + 8*hi + 0..7
  return u.v;
}

// B fragment: 32x16 (KxN) f16 with K-major (transposed) storage:
// element (k,n) at srcT[n*ld + k]. Per lane (n = lane&15, hi = lane>=16)
// K = [8*hi .. 8*hi+7] and [16+8*hi .. 16+8*hi+7] -> two 16-byte loads.
__device__ __forceinline__ v16h load_b_fragT(const h16* srcT, int ld) {
  const int lane = threadIdx.x & 31;
  const int n  = lane & 15;
  const int hi = (lane >> 4) & 1;
  const h16* p = srcT + (size_t)n * ld + hi * 8;
  union { v16h v; v8h h[2]; } u;
  u.h[0] = *reinterpret_cast<const v8h*>(p);        // K = 8*hi + 0..7
  u.h[1] = *reinterpret_cast<const v8h*>(p + 16);   // K = 16 + 8*hi + 0..7
  return u.v;
}

// ---------------------------------------------------------------------------
// GEMM: C = A(f32, MxK) @ W(f32, KxN) + bias
//  mode 0: out16[row*N+col] = (h16)C                      (row-major f16)
//  mode 1: outZ = resid + relu(C)                         (f32)
//  mode 2: out16 transposed [b, col, nk] (b = row/2048)   (f16, for V)
// Block: 256 threads (8 waves), tile 128(M) x 64(N), K step 32.
// ---------------------------------------------------------------------------
#define GEMM_BM 128
#define GEMM_BN 64
#define GEMM_BK 32
#define A_LD (GEMM_BK + 8)   // 40 halves = 80 B rows (16B-aligned)
#define BT_LD (GEMM_BK + 8)  // BsT rows (over K) = 40 halves

__global__ __launch_bounds__(256) void gemm_kernel(
    const float* __restrict__ A, const float* __restrict__ W,
    const float* __restrict__ bias, h16* __restrict__ out16,
    const float* __restrict__ resid, float* __restrict__ outZ,
    int M, int N, int K, int mode) {
  __shared__ alignas(16) h16 As[GEMM_BM][A_LD];
  __shared__ alignas(16) h16 BsT[GEMM_BN][BT_LD];  // [n][k] K-major

  const int tid    = threadIdx.x;
  const int wave   = tid >> 5;
  const int lane   = tid & 31;
  const int tile_m = blockIdx.x;
  const int tile_n = blockIdx.y;

  const float* Ab = A + (size_t)tile_m * GEMM_BM * K;
  const float* Bb = W + tile_n * GEMM_BN;

  v8f acc[4] = {v8f{}, v8f{}, v8f{}, v8f{}};

  for (int k0 = 0; k0 < K; k0 += GEMM_BK) {
    // stage A tile 128x32: 2048 float2 pairs, 8 per thread, packed stores
#pragma unroll
    for (int i = 0; i < 8; ++i) {
      int idx = tid + i * 256;          // pair index 0..2047
      int r = idx >> 4;                 // 16 pairs per row
      int c = (idx & 15) << 1;
      float2 f = *reinterpret_cast<const float2*>(&Ab[(size_t)r * K + k0 + c]);
      As[r][c]     = (h16)f.x;
      As[r][c + 1] = (h16)f.y;
    }
    // stage B tile 32x64 transposed into BsT[n][k] (global reads coalesced)
#pragma unroll
    for (int i = 0; i < 8; ++i) {
      int idx = tid + i * 256;          // 0..2047
      int r = idx >> 6, c = idx & 63;   // r = k, c = n
      BsT[c][r] = (h16)Bb[(size_t)(k0 + r) * N + c];
    }
    if (k0 + GEMM_BK < K) {  // gfx1250 global_prefetch for next K slab
      __builtin_prefetch(&Ab[(size_t)(tid >> 1) * K + k0 + GEMM_BK], 0, 1);
    }
    __syncthreads();

    v16h af = load_a_frag(&As[wave * 16][0], A_LD);
#pragma unroll
    for (int nf = 0; nf < 4; ++nf) {
      v16h bf = load_b_fragT(&BsT[nf * 16][0], BT_LD);
      acc[nf] = wmma_f16(af, bf, acc[nf]);
    }
    __syncthreads();
  }

  // epilogue: C layout -> row = 8*(lane>=16) + v, col = 16*nf + (lane&15)
  const int hi = (lane >> 4) & 1;
  const int n_lo = lane & 15;
  const int row0 = tile_m * GEMM_BM + wave * 16 + hi * 8;
#pragma unroll
  for (int nf = 0; nf < 4; ++nf) {
    int col = tile_n * GEMM_BN + nf * 16 + n_lo;
    float bval = bias[col];
    if (mode == 2) {
      // transposed f16 out: [b][col][nk], nk-dim = 2048; rows v are
      // consecutive nk -> one 16B packed store
      int bb  = row0 >> 11;
      int nk0 = row0 & 2047;
      v8h pack;
#pragma unroll
      for (int v = 0; v < 8; ++v) pack[v] = (h16)(acc[nf][v] + bval);
      *reinterpret_cast<v8h*>(
          &out16[((size_t)bb * 1024 + col) * 2048 + nk0]) = pack;
    } else {
#pragma unroll
      for (int v = 0; v < 8; ++v) {
        size_t idx = (size_t)(row0 + v) * N + col;
        float val = acc[nf][v] + bval;
        if (mode == 0) {
          out16[idx] = (h16)val;
        } else {
          outZ[idx] = resid[idx] + (val > 0.f ? val : 0.f);
        }
      }
    }
  }
}

// ---------------------------------------------------------------------------
// Flash attention per (b, h, 16-query tile). One wave per block.
// logits = q.k / sqrt(DV); online softmax; O = q + P@V (residual fused).
// K operand: k16 row-major over head-dim == K-major for B-frags (b128 loads).
// V operand: vt16 transposed [B][DV][NK] == K-major over keys (b128 loads).
// ---------------------------------------------------------------------------
#define ATT_DV 1024
#define ATT_DH 128
#define ATT_NQ 1024
#define ATT_NK 2048

__global__ __launch_bounds__(32) void attn_kernel(
    const h16* __restrict__ q16, const h16* __restrict__ k16,
    const h16* __restrict__ vt16, float* __restrict__ O) {
  const int id   = blockIdx.x;
  const int qt   = id & 63;         // NQ/16 = 64
  const int head = (id >> 6) & 7;   // H = 8
  const int b    = id >> 9;
  const int lane = threadIdx.x & 31;
  const int hi   = (lane >> 4) & 1;
  const int n_lo = lane & 15;
  const int hoff = head * ATT_DH;
  const size_t qrow0 = (size_t)b * ATT_NQ + qt * 16;

  __shared__ alignas(16) h16 Pl[16][32];  // P bounce: C-layout -> A-layout

  // Load 16x128 query tile as 4 A-fragments (K steps of 32 over head dim)
  v16h aq[4];
#pragma unroll
  for (int ks = 0; ks < 4; ++ks)
    aq[ks] = load_a_frag(q16 + qrow0 * ATT_DV + hoff + ks * 32, ATT_DV);

  v8f oacc[8] = {v8f{}, v8f{}, v8f{}, v8f{}, v8f{}, v8f{}, v8f{}, v8f{}};
  float m_i[8], l_i[8];
#pragma unroll
  for (int v = 0; v < 8; ++v) { m_i[v] = -__builtin_inff(); l_i[v] = 0.f; }

  const float scale = 0.03125f;  // 1/sqrt(1024)
  const h16* vbase = vt16 + ((size_t)b * ATT_DV + hoff) * ATT_NK;

  for (int j0 = 0; j0 < ATT_NK; j0 += 32) {
    const size_t krow0 = (size_t)b * ATT_NK + j0;

    // S(16x32) = q(16x128) @ K_tile^T(128x32), two 16x16 n-frags
    v8f s[2] = {v8f{}, v8f{}};
#pragma unroll
    for (int ks = 0; ks < 4; ++ks) {
#pragma unroll
      for (int nf = 0; nf < 2; ++nf) {
        v16h bk = load_b_fragT(
            k16 + (krow0 + nf * 16) * ATT_DV + hoff + ks * 32, ATT_DV);
        s[nf] = wmma_f16(aq[ks], bk, s[nf]);
      }
    }
#pragma unroll
    for (int nf = 0; nf < 2; ++nf)
#pragma unroll
      for (int v = 0; v < 8; ++v) s[nf][v] *= scale;

    // online softmax: per-row (row = 8*hi + v, spread over a 16-lane group)
    float alpha[8];
#pragma unroll
    for (int v = 0; v < 8; ++v) {
      float mx = fmaxf(s[0][v], s[1][v]);
      mx = fmaxf(mx, __shfl_xor(mx, 1));
      mx = fmaxf(mx, __shfl_xor(mx, 2));
      mx = fmaxf(mx, __shfl_xor(mx, 4));
      mx = fmaxf(mx, __shfl_xor(mx, 8));
      float mnew = fmaxf(m_i[v], mx);
      alpha[v] = __expf(m_i[v] - mnew);  // exp(-inf)=0 on first tile
      m_i[v] = mnew;
    }
#pragma unroll
    for (int nf = 0; nf < 2; ++nf)
#pragma unroll
      for (int v = 0; v < 8; ++v) s[nf][v] = __expf(s[nf][v] - m_i[v]);
#pragma unroll
    for (int v = 0; v < 8; ++v) {
      float rs = s[0][v] + s[1][v];
      rs += __shfl_xor(rs, 1);
      rs += __shfl_xor(rs, 2);
      rs += __shfl_xor(rs, 4);
      rs += __shfl_xor(rs, 8);
      l_i[v] = l_i[v] * alpha[v] + rs;
    }
#pragma unroll
    for (int nf = 0; nf < 8; ++nf)
#pragma unroll
      for (int v = 0; v < 8; ++v) oacc[nf][v] *= alpha[v];

    // C-layout P -> LDS -> A-layout fragment (two ds_load_b128 on read)
#pragma unroll
    for (int nf = 0; nf < 2; ++nf)
#pragma unroll
      for (int v = 0; v < 8; ++v)
        Pl[hi * 8 + v][nf * 16 + n_lo] = (h16)s[nf][v];
    __syncthreads();
    v16h ap = load_a_frag(&Pl[0][0], 32);

    // O(16x128) += P(16x32) @ V_tile(32x128): 8 n-frags over head dim
#pragma unroll
    for (int nf = 0; nf < 8; ++nf) {
      v16h bv = load_b_fragT(vbase + (size_t)(nf * 16) * ATT_NK + j0, ATT_NK);
      oacc[nf] = wmma_f16(ap, bv, oacc[nf]);
    }
    __syncthreads();
  }

  // epilogue: O = q + oacc / l  (residual fused)
#pragma unroll
  for (int nf = 0; nf < 8; ++nf) {
#pragma unroll
    for (int v = 0; v < 8; ++v) {
      int r = hi * 8 + v;
      size_t idx = (qrow0 + r) * ATT_DV + hoff + nf * 16 + n_lo;
      O[idx] = oacc[nf][v] * (1.0f / l_i[v]) + (float)q16[idx];
    }
  }
}

// ---------------------------------------------------------------------------
// Row LayerNorm over D=1024, one 256-thread block per row. In-place safe.
// ---------------------------------------------------------------------------
__global__ __launch_bounds__(256) void ln_kernel(
    const float* __restrict__ X, float* __restrict__ Y,
    const float* __restrict__ g, const float* __restrict__ bta, int D) {
  __shared__ float sbuf[8], ssbuf[8];
  const int row = blockIdx.x;
  const float* x = X + (size_t)row * D;
  float vals[4], s = 0.f, ss = 0.f;
#pragma unroll
  for (int i = 0; i < 4; ++i) {
    float v = x[threadIdx.x + i * 256];
    vals[i] = v;
    s += v;
    ss += v * v;
  }
#pragma unroll
  for (int m = 1; m < 32; m <<= 1) {
    s += __shfl_xor(s, m);
    ss += __shfl_xor(ss, m);
  }
  if ((threadIdx.x & 31) == 0) {
    sbuf[threadIdx.x >> 5] = s;
    ssbuf[threadIdx.x >> 5] = ss;
  }
  __syncthreads();
  s = 0.f; ss = 0.f;
#pragma unroll
  for (int w = 0; w < 8; ++w) { s += sbuf[w]; ss += ssbuf[w]; }
  float mu = s / (float)D;
  float var = ss / (float)D - mu * mu;
  float rstd = rsqrtf(var + 1e-5f);
#pragma unroll
  for (int i = 0; i < 4; ++i) {
    int c = threadIdx.x + i * 256;
    Y[(size_t)row * D + c] = (vals[i] - mu) * rstd * g[c] + bta[c];
  }
}

// ---------------------------------------------------------------------------
extern "C" void kernel_launch(void* const* d_in, const int* in_sizes, int n_in,
                              void* d_out, int out_size, void* d_ws,
                              size_t ws_size, hipStream_t stream) {
  const float* Q  = (const float*)d_in[0];
  const float* K  = (const float*)d_in[1];
  const float* Wq = (const float*)d_in[2];
  const float* bq = (const float*)d_in[3];
  const float* Wk = (const float*)d_in[4];
  const float* bk = (const float*)d_in[5];
  const float* Wv = (const float*)d_in[6];
  const float* bv = (const float*)d_in[7];
  const float* Wo = (const float*)d_in[8];
  const float* bo = (const float*)d_in[9];
  const float* g0 = (const float*)d_in[10];
  const float* b0 = (const float*)d_in[11];
  const float* g1 = (const float*)d_in[12];
  const float* b1 = (const float*)d_in[13];
  float* out = (float*)d_out;

  const int Bsz = 8, NQ = 1024, NK = 2048, D = 1024;
  const size_t nQ = (size_t)Bsz * NQ * D;  // 8M
  const size_t nK = (size_t)Bsz * NK * D;  // 16M

  char* ws = (char*)d_ws;
  h16* q16  = (h16*)ws;  ws += nQ * sizeof(h16);
  h16* k16  = (h16*)ws;  ws += nK * sizeof(h16);
  h16* vt16 = (h16*)ws;  ws += nK * sizeof(h16);   // [B][DV][NK] transposed
  float* O  = (float*)ws; ws += nQ * sizeof(float);
  float* Z  = (float*)ws; ws += nQ * sizeof(float);

  const int Mq = Bsz * NQ;   // 8192
  const int Mk = Bsz * NK;   // 16384

  // projections (f16 outputs for the attention core; V stored transposed)
  gemm_kernel<<<dim3(Mq / GEMM_BM, D / GEMM_BN), 256, 0, stream>>>(
      Q, Wq, bq, q16, nullptr, nullptr, Mq, D, D, 0);
  gemm_kernel<<<dim3(Mk / GEMM_BM, D / GEMM_BN), 256, 0, stream>>>(
      K, Wk, bk, k16, nullptr, nullptr, Mk, D, D, 0);
  gemm_kernel<<<dim3(Mk / GEMM_BM, D / GEMM_BN), 256, 0, stream>>>(
      K, Wv, bv, vt16, nullptr, nullptr, Mk, D, D, 2);

  // fused flash attention + residual
  attn_kernel<<<Bsz * 8 * (NQ / 16), 32, 0, stream>>>(q16, k16, vt16, O);

  // LN0 (in place)
  ln_kernel<<<Mq, 256, 0, stream>>>(O, O, g0, b0, D);

  // Z = O + relu(O @ Wo + bo)
  gemm_kernel<<<dim3(Mq / GEMM_BM, D / GEMM_BN), 256, 0, stream>>>(
      O, Wo, bo, nullptr, O, Z, Mq, D, D, 1);

  // LN1 -> out
  ln_kernel<<<Mq, 256, 0, stream>>>(Z, out, g1, b1, D);
}
